// NeighborhoodAttentionModule_6923487281850
// MI455X (gfx1250) — matrix operand order repaired and checked by reading, not compiled, not run on hardware
//
#include <hip/hip_runtime.h>

typedef __bf16 bf16;
typedef __attribute__((ext_vector_type(16))) __bf16 v16bf;
typedef __attribute__((ext_vector_type(8)))  __bf16 v8bf;
typedef __attribute__((ext_vector_type(8)))  float  v8f;

#define IN_DIM 256
#define NH 2
#define KN 16
#define TILE_B 16
#define AROWS (TILE_B * KN + TILE_B) /* 256 neighbor rows + 16 center rows = 272 */
#define STR 264                      /* row stride (bf16 elems): 528B, 16B-aligned, low bank conflict */

/* workspace (bf16) layout, element offsets */
#define WCC_OFF 0                    /* 256x256 : rows [W_center_out(128); W_center(128)] */
#define WNV_OFF (256 * 256)          /* 384x256 : rows [W_neighbor(128); W_value(256)]    */
#define WO_OFF  (WNV_OFF + 384 * 256)/* 128x256 : W_out                                   */
#define WTOT    (WO_OFF + 128 * 256)

#define WMMA_BF16(a, b, c) \
  __builtin_amdgcn_wmma_f32_16x16x32_bf16(false, (a), false, (b), (short)0, (c), false, false)

union BF4 { bf16 b[4]; uint2 u; };

/* --- convert all weight matrices fp32 -> bf16 into workspace, row-major --- */
__global__ void prep_weights(const float* __restrict__ Wc, const float* __restrict__ Wn,
                             const float* __restrict__ Wv, const float* __restrict__ Wo,
                             const float* __restrict__ Wco, bf16* __restrict__ wsb) {
  int idx = blockIdx.x * blockDim.x + threadIdx.x;
  if (idx >= WTOT) return;
  int row = idx >> 8, col = idx & 255;
  float v;
  if (row < 128)      v = Wco[row * 256 + col];
  else if (row < 256) v = Wc[(row - 128) * 256 + col];
  else if (row < 384) v = Wn[(row - 256) * 256 + col];
  else if (row < 640) v = Wv[(row - 384) * 256 + col];
  else                v = Wo[(row - 640) * 256 + col];
  wsb[idx] = (bf16)v;
}

/* A-fragment (16x32 bf16): lane lo<16 -> row M=lo; 16 contiguous K vals [k0+16hi .. +15] */
__device__ __forceinline__ v16bf load_a(const bf16* base, int rowoff, int k0, int hi) {
  v8bf c0 = *(const v8bf*)(base + rowoff + k0 + (hi << 4));
  v8bf c1 = *(const v8bf*)(base + rowoff + k0 + 8 + (hi << 4));
  v16bf a;
#pragma unroll
  for (int i = 0; i < 8; ++i) { a[i] = c0[i]; a[i + 8] = c1[i]; }
  return a;
}

/* B-fragment (32x16 bf16): lane holds output column (rowbase+lo) of W[N,256]; 16 contiguous K vals */
__device__ __forceinline__ v16bf load_b(const bf16* __restrict__ W, int rowbase, int lo, int hi, int k0) {
  return *(const v16bf*)(W + (size_t)(rowbase + lo) * IN_DIM + k0 + (hi << 4));
}

__launch_bounds__(512)
__global__ void natt_main(const float* __restrict__ center, const float* __restrict__ neigh,
                          const float* __restrict__ ew, const bf16* __restrict__ wsb,
                          float* __restrict__ out) {
  __shared__ bf16  sA[AROWS * STR];        /* rows 0..255 neighbors (b-major), 256..271 center */
  __shared__ bf16  sQ[TILE_B * 128];       /* q, bf16 */
  __shared__ float sS[TILE_B * NH * KN];   /* scores -> attn in place */
  __shared__ bf16  sC[TILE_B * STR];       /* context rows, bf16 */

  const int b0   = blockIdx.x * TILE_B;
  const int tid  = threadIdx.x;
  const int wv   = tid >> 5;      /* wave 0..15 */
  const int lane = tid & 31;
  const int lo   = lane & 15;
  const int hi   = lane >> 4;

  /* ---- stage activations fp32 -> bf16 into LDS ---- */
  {
    const float4* gn = (const float4*)(neigh + (size_t)b0 * KN * IN_DIM);
    for (int f = tid; f < (TILE_B * KN * IN_DIM) / 4; f += 512) {
      float4 x = gn[f];
      int row = f >> 6, col = (f & 63) << 2;
      BF4 p; p.b[0] = (bf16)x.x; p.b[1] = (bf16)x.y; p.b[2] = (bf16)x.z; p.b[3] = (bf16)x.w;
      *(uint2*)&sA[row * STR + col] = p.u;
    }
    const float4* gc = (const float4*)(center + (size_t)b0 * IN_DIM);
    for (int f = tid; f < (TILE_B * IN_DIM) / 4; f += 512) {
      float4 x = gc[f];
      int row = 256 + (f >> 6), col = (f & 63) << 2;
      BF4 p; p.b[0] = (bf16)x.x; p.b[1] = (bf16)x.y; p.b[2] = (bf16)x.z; p.b[3] = (bf16)x.w;
      *(uint2*)&sA[row * STR + col] = p.u;
    }
  }
  __syncthreads();

  /* ---- center GEMM: [16,256] x [256 -> 256 cols = center_out(128) | q(128)] ---- */
  {
    const bf16* Wcc = wsb + WCC_OFF;
    v8f acc = {};
    const int arow = (256 + lo) * STR;
#pragma unroll
    for (int ks = 0; ks < 8; ++ks) {
      const int k0 = ks * 32;
      v16bf a = load_a(sA, arow, k0, hi);
      v16bf b = load_b(Wcc, wv * 16, lo, hi, k0);
      acc = WMMA_BF16(a, b, acc);
    }
    if (wv < 8) {
#pragma unroll
      for (int r = 0; r < 8; ++r)
        out[(size_t)(b0 + r + 8 * hi) * IN_DIM + (wv * 16 + lo)] = acc[r];
    } else {
#pragma unroll
      for (int r = 0; r < 8; ++r)
        sQ[(r + 8 * hi) * 128 + ((wv - 8) * 16 + lo)] = (bf16)acc[r];
    }
  }
  __syncthreads();

  /* ---- per wave = batch element wv: k-projection fused into scores ---- */
  {
    const bf16* Wn = wsb + WNV_OFF;
    const int arow = (wv * 16 + lo) * STR;
    float p0[8], p1[8];
#pragma unroll
    for (int r = 0; r < 8; ++r) { p0[r] = 0.f; p1[r] = 0.f; }
#pragma unroll
    for (int nt = 0; nt < 8; ++nt) {
      v8f acc = {};
#pragma unroll
      for (int ks = 0; ks < 8; ++ks) {
        const int k0 = ks * 32;
        v16bf a = load_a(sA, arow, k0, hi);
        v16bf b = load_b(Wn, nt * 16, lo, hi, k0);
        acc = WMMA_BF16(a, b, acc);
      }
      float qa = (float)sQ[wv * 128 + nt * 16 + lo];
      if (nt < 4) {
#pragma unroll
        for (int r = 0; r < 8; ++r) p0[r] += qa * acc[r];
      } else {
#pragma unroll
        for (int r = 0; r < 8; ++r) p1[r] += qa * acc[r];
      }
    }
    /* reduce over 64 attn dims spread across the 16-lane group */
#pragma unroll
    for (int r = 0; r < 8; ++r) {
#pragma unroll
      for (int m = 1; m < 16; m <<= 1) {
        p0[r] += __shfl_xor(p0[r], m, 32);
        p1[r] += __shfl_xor(p1[r], m, 32);
      }
    }
    if (lo == 0) {
      const float scale = 0.125f; /* 64^-0.5 */
#pragma unroll
      for (int r = 0; r < 8; ++r) {
        int kn = r + 8 * hi;
        float e = ew[(size_t)(b0 + wv) * KN + kn];
        sS[(wv * NH + 0) * KN + kn] = p0[r] * scale + e;
        sS[(wv * NH + 1) * KN + kn] = p1[r] * scale + e;
      }
    }
  }
  __syncthreads();

  /* ---- softmax over K=16, one lane per (b,h) ---- */
  if (lane < NH) {
    float* s = &sS[(wv * NH + lane) * KN];
    float mx = -1e30f;
#pragma unroll
    for (int k = 0; k < KN; ++k) mx = fmaxf(mx, s[k]);
    float sum = 0.f, e[KN];
#pragma unroll
    for (int k = 0; k < KN; ++k) { e[k] = __expf(s[k] - mx); sum += e[k]; }
    float inv = 1.f / sum;
#pragma unroll
    for (int k = 0; k < KN; ++k) s[k] = e[k] * inv;
  }
  __syncthreads();

  /* ---- v-projection fused with attn contraction -> context row (bf16) ---- */
  {
    const bf16* Wvv = wsb + WNV_OFF + 128 * IN_DIM;
    const int arow = (wv * 16 + lo) * STR;
#pragma unroll
    for (int h = 0; h < NH; ++h) {
      float at[8];
#pragma unroll
      for (int r = 0; r < 8; ++r) at[r] = sS[(wv * NH + h) * KN + r + 8 * hi];
#pragma unroll
      for (int t = 0; t < 8; ++t) {
        const int nt = h * 8 + t;
        v8f acc = {};
#pragma unroll
        for (int ks = 0; ks < 8; ++ks) {
          const int k0 = ks * 32;
          v16bf a = load_a(sA, arow, k0, hi);
          v16bf b = load_b(Wvv, nt * 16, lo, hi, k0);
          acc = WMMA_BF16(a, b, acc);
        }
        float c = 0.f;
#pragma unroll
        for (int r = 0; r < 8; ++r) c += at[r] * acc[r];
        c += __shfl_xor(c, 16, 32);
        if (hi == 0) sC[wv * STR + nt * 16 + lo] = (bf16)c;
      }
    }
  }
  __syncthreads();

  /* ---- ctx_out GEMM: context[16,256] x W_out.T -> out cols 128..255 ---- */
  if (wv < 8) {
    const bf16* Wo = wsb + WO_OFF;
    v8f acc = {};
    const int arow = lo * STR;
#pragma unroll
    for (int ks = 0; ks < 8; ++ks) {
      const int k0 = ks * 32;
      v16bf a = load_a(sC, arow, k0, hi);
      v16bf b = load_b(Wo, wv * 16, lo, hi, k0);
      acc = WMMA_BF16(a, b, acc);
    }
#pragma unroll
    for (int r = 0; r < 8; ++r)
      out[(size_t)(b0 + r + 8 * hi) * IN_DIM + 128 + wv * 16 + lo] = acc[r];
  }
}

extern "C" void kernel_launch(void* const* d_in, const int* in_sizes, int n_in,
                              void* d_out, int out_size, void* d_ws, size_t ws_size,
                              hipStream_t stream) {
  const float* center = (const float*)d_in[0];
  const float* neigh  = (const float*)d_in[1];
  const float* ew     = (const float*)d_in[2];
  const float* Wc     = (const float*)d_in[3];
  const float* Wn     = (const float*)d_in[4];
  const float* Wv     = (const float*)d_in[5];
  const float* Wo     = (const float*)d_in[6];
  const float* Wco    = (const float*)d_in[7];
  float* outp = (float*)d_out;
  bf16* wsb   = (bf16*)d_ws;
  const int B = in_sizes[0] / IN_DIM;

  prep_weights<<<(WTOT + 255) / 256, 256, 0, stream>>>(Wc, Wn, Wv, Wo, Wco, wsb);
  natt_main<<<B / TILE_B, 512, 0, stream>>>(center, neigh, ew, wsb, outp);
}
